// FlaxGPTNeoXBlock_84619445666499
// MI455X (gfx1250) — matrix operand order repaired and checked by reading, not compiled, not run on hardware
//
#include <hip/hip_runtime.h>
#include <cstdint>
#include <cstddef>

// Problem constants (GPT-NeoX block)
#define B_   2
#define S_   2048
#define D_   2048
#define H_   16
#define HD_  128
#define DFF_ 8192

typedef __bf16 bf16_t;
typedef __bf16 bfx16 __attribute__((ext_vector_type(16)));
typedef float  fx8   __attribute__((ext_vector_type(8)));

struct alignas(16) B128 { unsigned int w[4]; };
union U16 { bfx16 v; B128 q[2]; };

static __device__ __forceinline__ fx8 wmma_bf16(bfx16 a, bfx16 b, fx8 c) {
  // D = A(16x32 bf16) * B(32x16 bf16) + C(16x16 f32)
  return __builtin_amdgcn_wmma_f32_16x16x32_bf16(false, a, false, b, (short)0, c,
                                                 false, false);
}

// CDNA5 async global->LDS copy (16B per lane), tracked by ASYNCcnt.
static __device__ __forceinline__ void async_load_b128(unsigned lds_off,
                                                       const bf16_t* g) {
  unsigned long long ga = (unsigned long long)(uintptr_t)g;
  asm volatile("global_load_async_to_lds_b128 %0, %1, off"
               :: "v"(lds_off), "v"(ga) : "memory");
}

// ---------------------------------------------------------------------------
// fp32 [K][N] -> bf16 [N][K] tiled transpose-convert (weights, every call)
// ---------------------------------------------------------------------------
__global__ __launch_bounds__(256) void cvt_transpose_bf16(
    const float* __restrict__ in, bf16_t* __restrict__ out, int K, int N)
{
  __shared__ bf16_t t[32][33];
  const int n0 = blockIdx.x * 32, k0 = blockIdx.y * 32;
  const int tx = threadIdx.x, ty = threadIdx.y;
#pragma unroll
  for (int i = 0; i < 4; i++)
    t[ty + 8 * i][tx] = (bf16_t)in[(size_t)(k0 + ty + 8 * i) * N + n0 + tx];
  __syncthreads();
#pragma unroll
  for (int i = 0; i < 4; i++)
    out[(size_t)(n0 + ty + 8 * i) * K + k0 + tx] = t[tx][ty + 8 * i];
}

// ---------------------------------------------------------------------------
// Fused LayerNorm1 + LayerNorm2 over rows of x[B*S, D] -> bf16 hs, hm
// ---------------------------------------------------------------------------
__global__ __launch_bounds__(256) void ln_fused(
    const float* __restrict__ x,
    const float* __restrict__ s1, const float* __restrict__ b1,
    const float* __restrict__ s2, const float* __restrict__ b2,
    bf16_t* __restrict__ hs, bf16_t* __restrict__ hm)
{
  __shared__ float red[256];
  const int row = blockIdx.x;
  const size_t base = (size_t)row * D_;
  float xr[8];
  float s = 0.f;
#pragma unroll
  for (int i = 0; i < 8; i++) { xr[i] = x[base + threadIdx.x + 256 * i]; s += xr[i]; }
  red[threadIdx.x] = s; __syncthreads();
  for (int o = 128; o > 0; o >>= 1) {
    if (threadIdx.x < o) red[threadIdx.x] += red[threadIdx.x + o];
    __syncthreads();
  }
  const float mu = red[0] * (1.f / D_);
  __syncthreads();
  float vv = 0.f;
#pragma unroll
  for (int i = 0; i < 8; i++) { float d = xr[i] - mu; vv += d * d; }
  red[threadIdx.x] = vv; __syncthreads();
  for (int o = 128; o > 0; o >>= 1) {
    if (threadIdx.x < o) red[threadIdx.x] += red[threadIdx.x + o];
    __syncthreads();
  }
  const float rs = rsqrtf(red[0] * (1.f / D_) + 1e-5f);
#pragma unroll
  for (int i = 0; i < 8; i++) {
    int c = threadIdx.x + 256 * i;
    float n = (xr[i] - mu) * rs;
    hs[base + c] = (bf16_t)(n * s1[c] + b1[c]);
    hm[base + c] = (bf16_t)(n * s2[c] + b2[c]);
  }
}

// ---------------------------------------------------------------------------
// WMMA bf16 GEMM: C[M,N] = A[M,K] * Bt[N,K]^T + bias
//   EPI=0: f32 out (+bias)  EPI=1: bf16 out (+bias)  EPI=2: bf16 (+bias,GELU)
// Block tile 128x128, K-step 32, 8 waves, each wave 2x4 WMMA 16x16 tiles.
// Both tiles land in LDS via double-buffered GLOBAL_LOAD_ASYNC_TO_LDS_B128
// (ASYNCcnt), so HBM latency overlaps WMMA issue with no VGPR staging.
// ---------------------------------------------------------------------------
template <int EPI>
__global__ __launch_bounds__(256) void gemm_bf16_wmma(
    const bf16_t* __restrict__ A, const bf16_t* __restrict__ Bt,
    const float* __restrict__ bias, void* __restrict__ Cout,
    int M, int N, int K)
{
  __shared__ __align__(16) bf16_t sA[2][128 * 32];
  __shared__ __align__(16) bf16_t sB[2][128 * 32]; // [n][k]
  const int tid = threadIdx.x;
  const int lane = tid & 31, wave = tid >> 5;
  const int wm = wave >> 1, wn = wave & 1;
  const int bm = blockIdx.y * 128, bn = blockIdx.x * 128;
  const int lr = lane & 15, lh = lane >> 4;
  const int nk = K >> 5;

  // Issue one 128x32 A tile + 128x32 B tile into LDS buffer `buf`
  // (4 async b128 instructions per wave per tile-pair).
  auto issue = [&](int kt, int buf) {
    const int k0 = kt << 5;
#pragma unroll
    for (int i = 0; i < 2; i++) {
      int c = tid + (i << 8);
      int r = c >> 2, ko = (c & 3) << 3;
      async_load_b128((unsigned)(uintptr_t)&sA[buf][r * 32 + ko],
                      A + (size_t)(bm + r) * K + k0 + ko);
      async_load_b128((unsigned)(uintptr_t)&sB[buf][r * 32 + ko],
                      Bt + (size_t)(bn + r) * K + k0 + ko);
    }
  };

  fx8 acc[2][4] = {};
  issue(0, 0);
  for (int kt = 0; kt < nk; ++kt) {
    const int cur = kt & 1;
    if (kt + 1 < nk) {
      issue(kt + 1, cur ^ 1);
      asm volatile("s_wait_asynccnt 0x4" ::: "memory"); // current tile landed
    } else {
      asm volatile("s_wait_asynccnt 0x0" ::: "memory");
    }
    __syncthreads();

    // A fragment (16x32): lanes 0-15 K 0..7 & 16..23; lanes 16-31 K 8..15 & 24..31
    bfx16 af[2], bfr[4];
#pragma unroll
    for (int ms = 0; ms < 2; ms++) {
      int r = wm * 32 + ms * 16 + lr, ko = lh << 3;
      U16 u;
      u.q[0] = *(const B128*)(&sA[cur][r * 32 + ko]);
      u.q[1] = *(const B128*)(&sA[cur][r * 32 + ko + 16]);
      af[ms] = u.v;
    }
    // B fragment (32x16): lane = column n; lanes 0-15 K 0..7 & 16..23; lanes 16-31 K 8..15 & 24..31
#pragma unroll
    for (int ns = 0; ns < 4; ns++) {
      int n = wn * 64 + ns * 16 + lr, ko = lh << 3;
      U16 u;
      u.q[0] = *(const B128*)(&sB[cur][n * 32 + ko]);
      u.q[1] = *(const B128*)(&sB[cur][n * 32 + ko + 16]);
      bfr[ns] = u.v;
    }
#pragma unroll
    for (int ms = 0; ms < 2; ms++)
#pragma unroll
      for (int ns = 0; ns < 4; ns++)
        acc[ms][ns] = wmma_bf16(af[ms], bfr[ns], acc[ms][ns]);
    __syncthreads();
  }

  // Epilogue. C/D layout: rows m = v + 8*lh, col n = lr.
#pragma unroll
  for (int ms = 0; ms < 2; ms++)
#pragma unroll
    for (int ns = 0; ns < 4; ns++) {
      int n = bn + wn * 64 + ns * 16 + lr;
      float bv = bias[n];
#pragma unroll
      for (int v = 0; v < 8; v++) {
        int m = bm + wm * 32 + ms * 16 + (lh << 3) + v;
        float val = acc[ms][ns][v] + bv;
        if (EPI == 2) {
          float x3 = val * val * val;
          val = 0.5f * val * (1.f + tanhf(0.7978845608028654f * (val + 0.044715f * x3)));
        }
        if (EPI == 0) ((float*)Cout)[(size_t)m * N + n] = val;
        else          ((bf16_t*)Cout)[(size_t)m * N + n] = (bf16_t)val;
      }
    }
}

// ---------------------------------------------------------------------------
// RoPE + QKV split. qkv[B*S, 3D] bf16 (bias already fused).
// q,k -> [B,H,S,HD]; v -> transposed [B,H,HD,S] (contiguous PV B-fragments).
// ---------------------------------------------------------------------------
__global__ __launch_bounds__(256) void rope_split(
    const bf16_t* __restrict__ qkv, bf16_t* __restrict__ qr,
    bf16_t* __restrict__ kr, bf16_t* __restrict__ vt)
{
  const int row = blockIdx.x;           // b*S + s
  const int b = row >> 11, s = row & (S_ - 1);
  const bf16_t* qk = qkv + (size_t)row * (3 * D_);
  const float kf = -logf(10000.f) / 64.f;
#pragma unroll
  for (int i = 0; i < 4; i++) {
    int p = threadIdx.x + (i << 8);     // pair index 0..1023
    int h = p >> 6, ii = p & 63;
    float ang = (float)s * expf(kf * (float)ii);
    float c = cosf(ang), sn = sinf(ang);
    int col = h * HD_ + 2 * ii;
    float q0 = (float)qk[col], q1 = (float)qk[col + 1];
    float k0 = (float)qk[D_ + col], k1 = (float)qk[D_ + col + 1];
    size_t o = (((size_t)b * H_ + h) * S_ + s) * HD_ + 2 * ii;
    qr[o]     = (bf16_t)(q0 * c - q1 * sn);
    qr[o + 1] = (bf16_t)(q0 * sn + q1 * c);
    kr[o]     = (bf16_t)(k0 * c - k1 * sn);
    kr[o + 1] = (bf16_t)(k0 * sn + k1 * c);
  }
#pragma unroll
  for (int i = 0; i < 8; i++) {
    int e = threadIdx.x + (i << 8);
    int h = e >> 7, d = e & 127;
    vt[(((size_t)b * H_ + h) * HD_ + d) * S_ + s] = qk[2 * D_ + e];
  }
}

// ---------------------------------------------------------------------------
// Flash attention, causal + additive mask. Grid (S/64, H, B), 128 threads.
// Per wave: 16 query rows; QK^T 4 WMMAs/16-key subtile; online softmax with
// 16-lane shuffle row reductions; P through per-wave LDS to A-fragment
// layout; PV 8 WMMAs per 32-key block.
// ---------------------------------------------------------------------------
__global__ __launch_bounds__(128) void attn_flash(
    const bf16_t* __restrict__ qr, const bf16_t* __restrict__ kr,
    const bf16_t* __restrict__ vt, const float* __restrict__ amask,
    bf16_t* __restrict__ ctx)
{
  __shared__ __align__(16) bf16_t sP[4][16 * 32];
  const int b = blockIdx.z, h = blockIdx.y;
  const int wave = threadIdx.x >> 5, lane = threadIdx.x & 31;
  const int lr = lane & 15, lh = lane >> 4;
  const int q0 = blockIdx.x * 64 + wave * 16;
  const bf16_t* qp = qr + ((size_t)b * H_ + h) * (size_t)S_ * HD_;
  const bf16_t* kp = kr + ((size_t)b * H_ + h) * (size_t)S_ * HD_;
  const bf16_t* vp = vt + ((size_t)b * H_ + h) * (size_t)HD_ * S_;
  const float* am = amask + (size_t)b * S_;
  bf16_t* myP = sP[wave];

  bfx16 qf[4];
#pragma unroll
  for (int kc = 0; kc < 4; kc++) {
    const bf16_t* a = qp + (size_t)(q0 + lr) * HD_ + kc * 32 + (lh << 3);
    U16 u; u.q[0] = *(const B128*)a; u.q[1] = *(const B128*)(a + 16);
    qf[kc] = u.v;
  }

  fx8 o[8] = {};
  float mi[8], li[8];
#pragma unroll
  for (int v = 0; v < 8; v++) { mi[v] = -1e30f; li[v] = 0.f; }
  const float rscale = 0.08838834764831845f; // 1/sqrt(128)

  const int nkb = (q0 + 16 + 31) >> 5;       // causal key-block count
  for (int ib = 0; ib < nkb; ++ib) {
    const int kb = ib << 5;
    fx8 st[2] = {};
#pragma unroll
    for (int t = 0; t < 2; t++) {
      int key = kb + t * 16 + lr; if (key > S_ - 1) key = S_ - 1; // clamp (masked)
      const bf16_t* kkp = kp + (size_t)key * HD_ + (lh << 3);
#pragma unroll
      for (int kc = 0; kc < 4; kc++) {
        U16 u;
        u.q[0] = *(const B128*)(kkp + kc * 32);
        u.q[1] = *(const B128*)(kkp + kc * 32 + 16);
        st[t] = wmma_bf16(qf[kc], u.v, st[t]);
      }
    }
    float ps[2][8];
#pragma unroll
    for (int v = 0; v < 8; v++) {
      int qrow = q0 + (lh << 3) + v;
#pragma unroll
      for (int t = 0; t < 2; t++) {
        int key = kb + t * 16 + lr;
        int keyc = key > S_ - 1 ? S_ - 1 : key;
        float sv = st[t][v] * rscale + am[keyc];
        st[t][v] = (key > qrow) ? -1e30f : sv;
      }
      float rm = fmaxf(st[0][v], st[1][v]);
#pragma unroll
      for (int off = 1; off < 16; off <<= 1) rm = fmaxf(rm, __shfl_xor(rm, off, 32));
      float mn = fmaxf(mi[v], rm);
      float al = expf(mi[v] - mn);
      float rs = 0.f;
#pragma unroll
      for (int t = 0; t < 2; t++) { ps[t][v] = expf(st[t][v] - mn); rs += ps[t][v]; }
#pragma unroll
      for (int off = 1; off < 16; off <<= 1) rs += __shfl_xor(rs, off, 32);
      li[v] = li[v] * al + rs;
      mi[v] = mn;
#pragma unroll
      for (int n = 0; n < 8; n++) o[n][v] *= al;
#pragma unroll
      for (int t = 0; t < 2; t++)
        myP[((lh << 3) + v) * 32 + t * 16 + lr] = (bf16_t)ps[t][v];
    }
    __threadfence_block(); // order per-wave ds stores before fragment reads

    U16 up;
    up.q[0] = *(const B128*)(&myP[lr * 32 + (lh << 4)]);
    up.q[1] = *(const B128*)(&myP[lr * 32 + (lh << 4) + 8]);
    bfx16 pf = up.v;

    int kbase = kb + (lh << 4); if (kbase > S_ - 16) kbase = S_ - 16; // p==0 there
#pragma unroll
    for (int n = 0; n < 8; n++) {
      const bf16_t* vv = vp + (size_t)(n * 16 + lr) * S_ + kbase;
      U16 u; u.q[0] = *(const B128*)vv; u.q[1] = *(const B128*)(vv + 8);
      o[n] = wmma_bf16(pf, u.v, o[n]);
    }
  }

#pragma unroll
  for (int v = 0; v < 8; v++) {
    int row = q0 + (lh << 3) + v;
    float inv = 1.f / li[v];
#pragma unroll
    for (int n = 0; n < 8; n++)
      ctx[((size_t)b * S_ + row) * D_ + h * HD_ + n * 16 + lr] = (bf16_t)(o[n][v] * inv);
  }
}

// ---------------------------------------------------------------------------
// out = x + attn_out + mlp_out (float4 vectorized)
// ---------------------------------------------------------------------------
struct alignas(16) F4 { float x, y, z, w; };
__global__ void final_add(const float* __restrict__ x, const float* __restrict__ a,
                          const float* __restrict__ m, float* __restrict__ out, int n4) {
  int i = blockIdx.x * blockDim.x + threadIdx.x;
  int stride = gridDim.x * blockDim.x;
  const F4* x4 = (const F4*)x; const F4* a4 = (const F4*)a; const F4* m4 = (const F4*)m;
  F4* o4 = (F4*)out;
  for (; i < n4; i += stride) {
    F4 xa = x4[i], aa = a4[i], ma = m4[i];
    F4 r; r.x = xa.x + aa.x + ma.x; r.y = xa.y + aa.y + ma.y;
    r.z = xa.z + aa.z + ma.z; r.w = xa.w + aa.w + ma.w;
    o4[i] = r;
  }
}

// ---------------------------------------------------------------------------
// Host launcher
// ---------------------------------------------------------------------------
extern "C" void kernel_launch(void* const* d_in, const int* in_sizes, int n_in,
                              void* d_out, int out_size, void* d_ws, size_t ws_size,
                              hipStream_t stream) {
  (void)in_sizes; (void)n_in; (void)out_size; (void)ws_size;
  const float* x    = (const float*)d_in[0];
  const float* am   = (const float*)d_in[1];
  const float* s1   = (const float*)d_in[2];
  const float* bb1  = (const float*)d_in[3];
  const float* s2   = (const float*)d_in[4];
  const float* bb2  = (const float*)d_in[5];
  const float* wqkv = (const float*)d_in[6];
  const float* bqkv = (const float*)d_in[7];
  const float* wo   = (const float*)d_in[8];
  const float* bo   = (const float*)d_in[9];
  const float* w1   = (const float*)d_in[10];
  const float* b1   = (const float*)d_in[11];
  const float* w2   = (const float*)d_in[12];
  const float* b2   = (const float*)d_in[13];

  char* ws = (char*)d_ws;
  constexpr size_t NT = (size_t)B_ * S_;             // 4096 token rows
  constexpr size_t OFF_HS   = 0;
  constexpr size_t OFF_HM   = OFF_HS   + NT * D_ * 2;
  constexpr size_t OFF_WQKV = OFF_HM   + NT * D_ * 2;
  constexpr size_t OFF_WO   = OFF_WQKV + (size_t)D_ * 3 * D_ * 2;
  constexpr size_t OFF_W1   = OFF_WO   + (size_t)D_ * D_ * 2;
  constexpr size_t OFF_W2   = OFF_W1   + (size_t)D_ * DFF_ * 2;
  constexpr size_t OFF_QKV  = OFF_W2   + (size_t)DFF_ * D_ * 2;
  constexpr size_t OFF_QR   = OFF_QKV  + NT * 3 * D_ * 2;
  constexpr size_t OFF_KR   = OFF_QR   + NT * D_ * 2;
  constexpr size_t OFF_VT   = OFF_KR   + NT * D_ * 2;
  constexpr size_t OFF_CTX  = OFF_VT   + NT * D_ * 2;
  constexpr size_t OFF_H1   = OFF_CTX  + NT * D_ * 2;
  constexpr size_t OFF_ATT  = OFF_H1   + NT * DFF_ * 2;
  constexpr size_t OFF_MLP  = OFF_ATT  + NT * D_ * 4;

  bf16_t* hs     = (bf16_t*)(ws + OFF_HS);
  bf16_t* hm     = (bf16_t*)(ws + OFF_HM);
  bf16_t* wqkv_t = (bf16_t*)(ws + OFF_WQKV);  // [3D][D]
  bf16_t* wo_t   = (bf16_t*)(ws + OFF_WO);    // [D][D]
  bf16_t* w1_t   = (bf16_t*)(ws + OFF_W1);    // [DFF][D]
  bf16_t* w2_t   = (bf16_t*)(ws + OFF_W2);    // [D][DFF]
  bf16_t* qkv_b  = (bf16_t*)(ws + OFF_QKV);
  bf16_t* q_r    = (bf16_t*)(ws + OFF_QR);
  bf16_t* k_r    = (bf16_t*)(ws + OFF_KR);
  bf16_t* v_t    = (bf16_t*)(ws + OFF_VT);
  bf16_t* ctx_b  = (bf16_t*)(ws + OFF_CTX);
  bf16_t* h1_b   = (bf16_t*)(ws + OFF_H1);
  float*  att_f  = (float*)(ws + OFF_ATT);
  float*  mlp_f  = (float*)(ws + OFF_MLP);

  const dim3 tb(32, 8);
  // 1) weights -> bf16, pre-transposed to [N][K]
  cvt_transpose_bf16<<<dim3(3 * D_ / 32, D_ / 32), tb, 0, stream>>>(wqkv, wqkv_t, D_, 3 * D_);
  cvt_transpose_bf16<<<dim3(D_ / 32, D_ / 32), tb, 0, stream>>>(wo, wo_t, D_, D_);
  cvt_transpose_bf16<<<dim3(DFF_ / 32, D_ / 32), tb, 0, stream>>>(w1, w1_t, D_, DFF_);
  cvt_transpose_bf16<<<dim3(D_ / 32, DFF_ / 32), tb, 0, stream>>>(w2, w2_t, DFF_, D_);
  // 2) fused dual layernorm
  ln_fused<<<(int)NT, 256, 0, stream>>>(x, s1, bb1, s2, bb2, hs, hm);
  // 3) QKV projection (bias fused, bf16 out)
  gemm_bf16_wmma<1><<<dim3(3 * D_ / 128, NT / 128), 256, 0, stream>>>(
      hs, wqkv_t, bqkv, qkv_b, (int)NT, 3 * D_, D_);
  // 4) RoPE + split (+ V transpose)
  rope_split<<<(int)NT, 256, 0, stream>>>(qkv_b, q_r, k_r, v_t);
  // 5) flash attention
  attn_flash<<<dim3(S_ / 64, H_, B_), 128, 0, stream>>>(q_r, k_r, v_t, am, ctx_b);
  // 6) output projection -> f32
  gemm_bf16_wmma<0><<<dim3(D_ / 128, NT / 128), 256, 0, stream>>>(
      ctx_b, wo_t, bo, att_f, (int)NT, D_, D_);
  // 7) MLP up + GELU -> bf16
  gemm_bf16_wmma<2><<<dim3(DFF_ / 128, NT / 128), 256, 0, stream>>>(
      hm, w1_t, b1, h1_b, (int)NT, DFF_, D_);
  // 8) MLP down -> f32
  gemm_bf16_wmma<0><<<dim3(D_ / 128, NT / 128), 256, 0, stream>>>(
      h1_b, w2_t, b2, mlp_f, (int)NT, D_, DFF_);
  // 9) parallel residual sum
  final_add<<<2048, 256, 0, stream>>>(x, att_f, mlp_f, (float*)d_out, (int)(NT * D_ / 4));
}
